// EdgeConv_69810398429321
// MI455X (gfx1250) — matrix operand order repaired and checked by reading, not compiled, not run on hardware
//
#include <hip/hip_runtime.h>
#include <stdint.h>

typedef __attribute__((ext_vector_type(16))) _Float16 v16h;
typedef __attribute__((ext_vector_type(8)))  _Float16 v8h;
typedef __attribute__((ext_vector_type(4)))  _Float16 v4h;
typedef __attribute__((ext_vector_type(8)))  float    v8f;

#define BATCH 8
#define NPTS  4096
#define CIN   64
#define COUT  64
#define KNBR  16
// Padded LDS row stride in halves: 136 halves = 272 B = 68 banks -> consecutive
// rows shift by 4 banks, so 16 row-parallel ds_load_b128 are conflict-free.
#define WSTRIDE 136
#define ASTRIDE 136

__device__ __forceinline__ v16h concat8(v8h lo, v8h hi) {
    return __builtin_shufflevector(lo, hi, 0,1,2,3,4,5,6,7,8,9,10,11,12,13,14,15);
}

__global__ __launch_bounds__(256)
void edgeconv_wmma_kernel(const float* __restrict__ x,
                          const long long* __restrict__ edge_index,
                          const float* __restrict__ weight,
                          const float* __restrict__ bias,
                          float* __restrict__ out)
{
    __shared__ _Float16 W_lds[COUT * WSTRIDE];        // 64 x 128 f16 (padded rows)
    __shared__ _Float16 A_lds[8][KNBR * ASTRIDE];     // per-wave 16 x 128 f16 (padded)

    const int tid  = threadIdx.x;
    const int lane = tid & 31;
    const int wave = tid >> 5;

    // ---- stage weight (f32 -> f16) into LDS once per workgroup: [o][c] ----
    {
        const float4* w4 = (const float4*)weight;
        for (int ci = tid; ci < (COUT * 2 * CIN) / 4; ci += 256) {
            int o  = ci >> 5;              // 32 float4 per 128-wide row
            int cq = (ci & 31) << 2;
            float4 w = w4[ci];
            v4h p = { (_Float16)w.x, (_Float16)w.y, (_Float16)w.z, (_Float16)w.w };
            *(v4h*)(&W_lds[o * WSTRIDE + cq]) = p;
        }
    }
    __syncthreads();

    // ---- one node per wave ----
    const int gid = blockIdx.x * 8 + wave;             // 0 .. B*N-1
    const int b   = gid >> 12;                         // / 4096
    const int n   = gid & (NPTS - 1);

    // ---- stage A = [x_central(64) | x_nbr - x_central(64)] per neighbor ----
    {
        const int r = lane >> 1;                       // neighbor row 0..15
        const int g = lane & 1;                        // column half 0..1
        const long long nb = edge_index[((size_t)b * NPTS + n) * KNBR + r];
        const float4* xc4 = (const float4*)(x + ((size_t)b * NPTS + n)  * CIN);
        const float4* xn4 = (const float4*)(x + ((size_t)b * NPTS + (size_t)nb) * CIN);
        _Float16* arow = &A_lds[wave][r * ASTRIDE];
        #pragma unroll
        for (int j = 0; j < 8; ++j) {
            int c = g * 32 + j * 4;                    // 0..63 in float4 steps
            float4 fc = xc4[c >> 2];
            float4 fn = xn4[c >> 2];
            v4h pc = { (_Float16)fc.x, (_Float16)fc.y, (_Float16)fc.z, (_Float16)fc.w };
            v4h pd = { (_Float16)(fn.x - fc.x), (_Float16)(fn.y - fc.y),
                       (_Float16)(fn.z - fc.z), (_Float16)(fn.w - fc.w) };
            *(v4h*)(arow + c)       = pc;              // central -> cols 0..63
            *(v4h*)(arow + CIN + c) = pd;              // diff    -> cols 64..127
        }
    }
    // A region is wave-private; DS ops from one wave complete in order
    // (compiler inserts s_wait_dscnt for the RAW below).

    // ---- GEMM: [16 x 128] x [128 x 64] via 16x v_wmma_f32_16x16x32_f16 ----
    v8f acc[4] = {v8f{}, v8f{}, v8f{}, v8f{}};
    const _Float16* arow = &A_lds[wave][(lane & 15) * ASTRIDE];
    const int ahalf = ((lane >> 4) & 1) * 8;           // A: half-wave K skew
    const int bhalf = ((lane >> 4) & 1) * 8;           // B: half-wave K skew
    #pragma unroll
    for (int ks = 0; ks < 4; ++ks) {
        const int abase = ks * 32 + ahalf;
        v16h a = concat8(*(const v8h*)(arow + abase),
                         *(const v8h*)(arow + abase + 16));
        const int kb = ks * 32 + bhalf;
        #pragma unroll
        for (int t = 0; t < 4; ++t) {
            const _Float16* wrow = &W_lds[(t * 16 + (lane & 15)) * WSTRIDE + kb];
            v16h bm = concat8(*(const v8h*)(wrow), *(const v8h*)(wrow + 16));
            acc[t] = __builtin_amdgcn_wmma_f32_16x16x32_f16(
                         false, a, false, bm, (short)0, acc[t], false, false);
        }
    }

    // ---- max over the 16 neighbor rows, bias + relu, transposed store ----
    // C/D layout: VGPR r holds M=r (lanes 0-15) and M=r+8 (lanes 16-31), N=lane&15.
    float* outb = out + (size_t)b * COUT * NPTS + n;
    #pragma unroll
    for (int t = 0; t < 4; ++t) {
        float m = acc[t][0];
        #pragma unroll
        for (int i = 1; i < 8; ++i) m = fmaxf(m, acc[t][i]);
        m = fmaxf(m, __shfl_xor(m, 16, 32));           // merge M=0..7 with M=8..15
        if (lane < 16) {
            int o = t * 16 + lane;
            float r = m + bias[o];                     // max_k relu(z+b) == relu(max_k z + b)
            outb[(size_t)o * NPTS] = r > 0.0f ? r : 0.0f;
        }
    }
}

extern "C" void kernel_launch(void* const* d_in, const int* in_sizes, int n_in,
                              void* d_out, int out_size, void* d_ws, size_t ws_size,
                              hipStream_t stream) {
    const float*     xp = (const float*)d_in[0];
    const long long* ei = (const long long*)d_in[1];   // int64 edge_index
    const float*     wp = (const float*)d_in[2];
    const float*     bp = (const float*)d_in[3];
    float*          outp = (float*)d_out;
    dim3 grid(BATCH * NPTS / 8);                       // 8 waves (nodes) per block
    dim3 block(256);
    hipLaunchKernelGGL(edgeconv_wmma_kernel, grid, block, 0, stream,
                       xp, ei, wp, bp, outp);
}
